// SpanPairPairedLayer_63376537420533
// MI455X (gfx1250) — compile-verified, matrix operand order checked by reading, not run on hardware
//
#include <hip/hip_runtime.h>
#include <hip/hip_bf16.h>

// ---------------------------------------------------------------------------
// SpanPairPairedLayer for MI455X (gfx1250, wave32, WMMA)
//
// prep:   span f32 -> bf16 (linear);  W1,Wr f32 -> bf16 transposed (n-major)
// GEMM1:  span_bf(8192x1024) x W1 -> relu -> span1 bf16     (8.6 GFLOP)
// GEMM2:  gather-concat pairs (32768x1024) x Wr -> relu f32 (34.4 GFLOP)
//
// GEMM blocks: 256 threads = 8 waves, tile 256M x 64N, K-step 32.
// Each wave: 32M x 64N = 8 x v_wmma_f32_16x16x32_bf16 per K-step,
// fragments via conflict-free ds_load_b128 from padded LDS tiles.
// ---------------------------------------------------------------------------

typedef __attribute__((ext_vector_type(16))) __bf16 v16bf;
typedef __attribute__((ext_vector_type(8)))  __bf16 v8bf;
typedef __attribute__((ext_vector_type(8)))  float  v8f;
typedef __attribute__((ext_vector_type(4)))  float  v4f;

// Padded LDS row stride for a 32-wide K tile: 40 elems = 80B.
// 16B-aligned (b128-legal); 20-bank stride -> 16 lanes hit 16 distinct
// 4-bank groups: conflict-free ds_load_b128 fragment reads.
#define LDA 40

#define S_LEN 1024
#define D_IN  1024
#define DR    512
#define DO    512
#define P_CNT 4096

// ISA 7.12.2 (16-bit A/B fragment, 16x32): lanes 0-15 hold K=kb..kb+7
// (VGPR0-3) and K=16+kb..16+kb+7 (VGPR4-7) with kb=0; lanes 16-31 use kb=8.
__device__ inline v16bf frag_from_lds(const __bf16* rowBase, int kb) {
    v8bf lo = *(const v8bf*)(rowBase + kb);
    v8bf hi = *(const v8bf*)(rowBase + 16 + kb);
    return __builtin_shufflevector(lo, hi, 0, 1, 2, 3, 4, 5, 6, 7,
                                           8, 9, 10, 11, 12, 13, 14, 15);
}

// ========================= prep kernels ====================================

// span f32 -> bf16, linear copy. 8 elems / thread.
__global__ __launch_bounds__(256) void conv_span_bf16(
    const float* __restrict__ in, __bf16* __restrict__ out)
{
    size_t i = ((size_t)blockIdx.x * 256 + threadIdx.x) * 8;
    v4f a = *(const v4f*)(in + i);
    v4f b = *(const v4f*)(in + i + 4);
    v8bf o;
    o[0] = (__bf16)a[0]; o[1] = (__bf16)a[1];
    o[2] = (__bf16)a[2]; o[3] = (__bf16)a[3];
    o[4] = (__bf16)b[0]; o[5] = (__bf16)b[1];
    o[6] = (__bf16)b[2]; o[7] = (__bf16)b[3];
    *(v8bf*)(out + i) = o;
}

// W (K=1024 x N=512 f32, row-major) -> Wt (N x K bf16), LDS tile transpose.
__global__ __launch_bounds__(256) void conv_transpose_w(
    const float* __restrict__ W, __bf16* __restrict__ Wt)
{
    __shared__ float tile[32][33];
    const int k0 = blockIdx.x * 32;
    const int n0 = blockIdx.y * 32;
    const int tx = threadIdx.x & 31;
    const int ty = threadIdx.x >> 5;     // 8 rows of threads
    #pragma unroll
    for (int j = 0; j < 4; ++j) {
        int k = ty + j * 8;
        tile[k][tx] = W[(size_t)(k0 + k) * DR + n0 + tx];   // coalesced in n
    }
    __syncthreads();
    #pragma unroll
    for (int j = 0; j < 4; ++j) {
        int n = ty + j * 8;
        Wt[(size_t)(n0 + n) * D_IN + k0 + tx] = (__bf16)tile[tx][n]; // coalesced in k
    }
}

// ============================ GEMM 1 =======================================
__global__ __launch_bounds__(256) void spanpair_gemm1(
    const __bf16* __restrict__ span_bf, // (8192, 1024) bf16
    const __bf16* __restrict__ Wt1,     // (512, 1024)  bf16 (n-major)
    const float*  __restrict__ b1,      // (512,)
    __bf16* __restrict__ span1)         // (8192, 512) bf16 out
{
    __shared__ __bf16 As[256 * LDA];    // 256 rows x 32 K (padded)
    __shared__ __bf16 Bt[64  * LDA];    // 64 cols  x 32 K

    const int t    = threadIdx.x;
    const int lane = t & 31;
    const int w    = t >> 5;
    const int m0   = blockIdx.x * 256;
    const int n0   = blockIdx.y * 64;

    v8f acc[8] = {};

    const int kb   = (lane >> 4) << 3;          // 0 | 8
    const int arow = w * 32 + (lane & 15);      // first of this lane's 2 A rows
    const int bn   = lane & 15;

    for (int k0 = 0; k0 < D_IN; k0 += 32) {
        __syncthreads();

        // ---- stage A: 256x32 bf16, pure b128 copies (4 per thread) ----
        #pragma unroll
        for (int it = 0; it < 4; ++it) {
            int i   = t + 256 * it;
            int row = i >> 2;
            int kc  = i & 3;
            const __bf16* src = span_bf + (size_t)(m0 + row) * D_IN + k0 + kc * 8;
            *(v8bf*)&As[row * LDA + kc * 8] = *(const v8bf*)src;
            __builtin_prefetch(src + 32, 0, 0);    // next K tile
        }
        // ---- stage B: 64x32 bf16 (already n-major), 1 b128 per thread ----
        {
            int n  = t >> 2;
            int kc = t & 3;
            const __bf16* src = Wt1 + (size_t)(n0 + n) * D_IN + k0 + kc * 8;
            *(v8bf*)&Bt[n * LDA + kc * 8] = *(const v8bf*)src;
            __builtin_prefetch(src + 32, 0, 0);
        }
        __syncthreads();

        // ---- 6 independent fragment loads, then 8 WMMAs ----
        v16bf a0 = frag_from_lds(&As[arow * LDA], kb);
        v16bf a1 = frag_from_lds(&As[(arow + 16) * LDA], kb);
        v16bf b0 = frag_from_lds(&Bt[bn * LDA],        kb);
        v16bf b1 = frag_from_lds(&Bt[(bn + 16) * LDA], kb);
        v16bf b2 = frag_from_lds(&Bt[(bn + 32) * LDA], kb);
        v16bf b3 = frag_from_lds(&Bt[(bn + 48) * LDA], kb);
        acc[0] = __builtin_amdgcn_wmma_f32_16x16x32_bf16(false, a0, false, b0, (short)0, acc[0], false, false);
        acc[1] = __builtin_amdgcn_wmma_f32_16x16x32_bf16(false, a0, false, b1, (short)0, acc[1], false, false);
        acc[2] = __builtin_amdgcn_wmma_f32_16x16x32_bf16(false, a0, false, b2, (short)0, acc[2], false, false);
        acc[3] = __builtin_amdgcn_wmma_f32_16x16x32_bf16(false, a0, false, b3, (short)0, acc[3], false, false);
        acc[4] = __builtin_amdgcn_wmma_f32_16x16x32_bf16(false, a1, false, b0, (short)0, acc[4], false, false);
        acc[5] = __builtin_amdgcn_wmma_f32_16x16x32_bf16(false, a1, false, b1, (short)0, acc[5], false, false);
        acc[6] = __builtin_amdgcn_wmma_f32_16x16x32_bf16(false, a1, false, b2, (short)0, acc[6], false, false);
        acc[7] = __builtin_amdgcn_wmma_f32_16x16x32_bf16(false, a1, false, b3, (short)0, acc[7], false, false);
    }

    // ---- epilogue: bias + relu -> bf16 (C/D layout: VGPR j = row j | j+8)
    #pragma unroll
    for (int mt = 0; mt < 2; ++mt) {
        int mbase = m0 + w * 32 + mt * 16 + ((lane >> 4) << 3);
        #pragma unroll
        for (int tn = 0; tn < 4; ++tn) {
            int n = n0 + tn * 16 + (lane & 15);
            float bias = b1[n];
            #pragma unroll
            for (int j = 0; j < 8; ++j) {
                float v = acc[mt * 4 + tn][j] + bias;
                v = v > 0.0f ? v : 0.0f;
                span1[(size_t)(mbase + j) * DR + n] = (__bf16)v;
            }
        }
    }
}

// ============================ GEMM 2 =======================================
__global__ __launch_bounds__(256) void spanpair_gemm2(
    const __bf16* __restrict__ span1, // (8*1024, 512) bf16
    const int*    __restrict__ pairs, // (8, 4096, 2) int32
    const __bf16* __restrict__ Wtr,   // (512, 1024) bf16 (n-major)
    const float*  __restrict__ br,    // (512,)
    float* __restrict__ out)          // (8*4096, 512) f32
{
    __shared__ __bf16 As[256 * LDA];
    __shared__ __bf16 Bt[64  * LDA];
    __shared__ int rowP[256];
    __shared__ int rowC[256];

    const int t    = threadIdx.x;
    const int lane = t & 31;
    const int w    = t >> 5;
    const int g0   = blockIdx.x * 256;   // pair-row base (0..32767)
    const int n0   = blockIdx.y * 64;

    // gather indices -> absolute span1 rows (one per thread)
    {
        int g = g0 + t;
        int b = g >> 12;                 // / 4096
        int p = g & (P_CNT - 1);
        const int* pr = pairs + ((size_t)b * P_CNT + p) * 2;
        rowP[t] = b * S_LEN + pr[0];
        rowC[t] = b * S_LEN + pr[1];
    }

    v8f acc[8] = {};
    const int kb   = (lane >> 4) << 3;
    const int arow = w * 32 + (lane & 15);
    const int bn   = lane & 15;

    for (int k0 = 0; k0 < 2 * DR; k0 += 32) {
        __syncthreads();
        const int* rowsel = (k0 < DR) ? rowP : rowC;  // concat half
        const int  ko     = k0 & (DR - 1);

        // ---- stage gathered A: 256x32 bf16, b128 copies ----
        #pragma unroll
        for (int it = 0; it < 4; ++it) {
            int i   = t + 256 * it;
            int row = i >> 2;
            int kc  = i & 3;
            const __bf16* src = span1 + (size_t)rowsel[row] * DR + ko + kc * 8;
            *(v8bf*)&As[row * LDA + kc * 8] = *(const v8bf*)src;
        }
        // ---- stage B: Wtr tile, 1 b128 per thread ----
        {
            int n  = t >> 2;
            int kc = t & 3;
            const __bf16* src = Wtr + (size_t)(n0 + n) * D_IN + k0 + kc * 8;
            *(v8bf*)&Bt[n * LDA + kc * 8] = *(const v8bf*)src;
            __builtin_prefetch(src + 32, 0, 0);
        }
        __syncthreads();

        v16bf a0 = frag_from_lds(&As[arow * LDA], kb);
        v16bf a1 = frag_from_lds(&As[(arow + 16) * LDA], kb);
        v16bf b0 = frag_from_lds(&Bt[bn * LDA],        kb);
        v16bf b1 = frag_from_lds(&Bt[(bn + 16) * LDA], kb);
        v16bf b2 = frag_from_lds(&Bt[(bn + 32) * LDA], kb);
        v16bf b3 = frag_from_lds(&Bt[(bn + 48) * LDA], kb);
        acc[0] = __builtin_amdgcn_wmma_f32_16x16x32_bf16(false, a0, false, b0, (short)0, acc[0], false, false);
        acc[1] = __builtin_amdgcn_wmma_f32_16x16x32_bf16(false, a0, false, b1, (short)0, acc[1], false, false);
        acc[2] = __builtin_amdgcn_wmma_f32_16x16x32_bf16(false, a0, false, b2, (short)0, acc[2], false, false);
        acc[3] = __builtin_amdgcn_wmma_f32_16x16x32_bf16(false, a0, false, b3, (short)0, acc[3], false, false);
        acc[4] = __builtin_amdgcn_wmma_f32_16x16x32_bf16(false, a1, false, b0, (short)0, acc[4], false, false);
        acc[5] = __builtin_amdgcn_wmma_f32_16x16x32_bf16(false, a1, false, b1, (short)0, acc[5], false, false);
        acc[6] = __builtin_amdgcn_wmma_f32_16x16x32_bf16(false, a1, false, b2, (short)0, acc[6], false, false);
        acc[7] = __builtin_amdgcn_wmma_f32_16x16x32_bf16(false, a1, false, b3, (short)0, acc[7], false, false);
    }

    // ---- epilogue: bias + relu -> f32 ----
    #pragma unroll
    for (int mt = 0; mt < 2; ++mt) {
        int gbase = g0 + w * 32 + mt * 16 + ((lane >> 4) << 3);
        #pragma unroll
        for (int tn = 0; tn < 4; ++tn) {
            int n = n0 + tn * 16 + (lane & 15);
            float bias = br[n];
            #pragma unroll
            for (int j = 0; j < 8; ++j) {
                float v = acc[mt * 4 + tn][j] + bias;
                v = v > 0.0f ? v : 0.0f;
                out[(size_t)(gbase + j) * DO + n] = v;
            }
        }
    }
}

// ============================ launch =======================================
extern "C" void kernel_launch(void* const* d_in, const int* in_sizes, int n_in,
                              void* d_out, int out_size, void* d_ws, size_t ws_size,
                              hipStream_t stream) {
    const float* span  = (const float*)d_in[0];
    const int*   pairs = (const int*)d_in[1];
    const float* W1    = (const float*)d_in[2];
    const float* b1    = (const float*)d_in[3];
    const float* Wr    = (const float*)d_in[4];
    const float* br    = (const float*)d_in[5];
    float*       outp  = (float*)d_out;

    // workspace layout (bytes):
    //   [0,           16M)  span_bf : 8192*1024 bf16
    //   [16M,         24M)  span1   : 8192*512  bf16
    //   [24M,         25M)  Wt1     : 512*1024  bf16
    //   [25M,         26M)  Wtr     : 512*1024  bf16
    char* ws = (char*)d_ws;
    __bf16* span_bf = (__bf16*)(ws);
    __bf16* span1   = (__bf16*)(ws + (size_t)16 * 1024 * 1024);
    __bf16* Wt1     = (__bf16*)(ws + (size_t)24 * 1024 * 1024);
    __bf16* Wtr     = (__bf16*)(ws + (size_t)25 * 1024 * 1024);

    dim3 blk(256, 1, 1);

    // prep
    conv_span_bf16<<<dim3((8192u * 1024u) / (256u * 8u), 1, 1), blk, 0, stream>>>(span, span_bf);
    conv_transpose_w<<<dim3(D_IN / 32, DR / 32, 1), blk, 0, stream>>>(W1, Wt1);
    conv_transpose_w<<<dim3(D_IN / 32, DO / 32, 1), blk, 0, stream>>>(Wr, Wtr);

    // GEMM1: 8192 x 512 x 1024
    spanpair_gemm1<<<dim3(8192 / 256, DR / 64, 1), blk, 0, stream>>>(span_bf, Wt1, b1, span1);

    // GEMM2: 32768 x 512 x 1024 (fused gather)
    spanpair_gemm2<<<dim3(32768 / 256, DO / 64, 1), blk, 0, stream>>>(span1, pairs, Wtr, br, outp);
}
